// S4DNeuroModel_12850542149719
// MI455X (gfx1250) — matrix operand order repaired and checked by reading, not compiled, not run on hardware
//
#include <hip/hip_runtime.h>
#include <hip/hip_bf16.h>

// ---- problem constants ----
#define Bsz  32
#define Lseq 2048
#define CIN  192
#define HD   512
#define NST  32      // N2 complex modes
#define NLAY 4
#define NCH  8       // scan chunks
#define CLEN 256     // Lseq / NCH

typedef __attribute__((ext_vector_type(16))) _Float16 v16h;
typedef __attribute__((ext_vector_type(8)))  float    v8f;
typedef int v4i __attribute__((vector_size(16)));

union V16H { v16h v; uint4 q[2]; };

// ---------------- async copy to LDS (CDNA5 path, guarded) ----------------
#if __has_builtin(__builtin_amdgcn_global_load_async_to_lds_b128)
#define HAVE_ASYNC 1
#else
#define HAVE_ASYNC 0
#endif

__device__ inline void async_cp16(void* lds, const void* g) {
#if HAVE_ASYNC
  __builtin_amdgcn_global_load_async_to_lds_b128(
      (__attribute__((address_space(1))) v4i*)g,
      (__attribute__((address_space(3))) v4i*)lds, 0, 0);
#else
  *(uint4*)lds = *(const uint4*)g;
#endif
}

// wait until this wave's outstanding async loads <= 0 / <= 2
__device__ inline void async_wait_le0() {
#if HAVE_ASYNC
#if __has_builtin(__builtin_amdgcn_s_wait_asynccnt)
  __builtin_amdgcn_s_wait_asynccnt(0);
#else
  asm volatile("s_wait_asynccnt 0" ::: "memory");
#endif
#endif
}
__device__ inline void async_wait_le2() {
#if HAVE_ASYNC
#if __has_builtin(__builtin_amdgcn_s_wait_asynccnt)
  __builtin_amdgcn_s_wait_asynccnt(2);
#else
  asm volatile("s_wait_asynccnt 2" ::: "memory");
#endif
#endif
}

// ---------------- WMMA helper ----------------
__device__ inline v8f wmma16(v16h a, v16h b, v8f c) {
  return __builtin_amdgcn_wmma_f32_16x16x32_f16(
      false, a, false, b, (short)0, c, false, false);
}

// A-frag from LDS row (k-chunk local): element j -> k = 16*hi + j
__device__ inline v16h a_frag(const _Float16* row, int hi) {
  V16H r;
  r.q[0] = *(const uint4*)(row + 16 * hi);
  r.q[1] = *(const uint4*)(row + 16 * hi + 8);
  return r.v;
}
// B-frag from global weight row (k-contiguous): element j -> k = 16*hi + j
__device__ inline v16h b_frag(const _Float16* p, int hi) {
  V16H r;
  r.q[0] = *(const uint4*)(p + 16 * hi);
  r.q[1] = *(const uint4*)(p + 16 * hi + 8);
  return r.v;
}

// ---------------- casts ----------------
__global__ void k_cast_f16(const float* __restrict__ s, _Float16* __restrict__ d, int n) {
  int i = blockIdx.x * 256 + threadIdx.x;
  if (i < n) d[i] = (_Float16)s[i];
}
// W_in (CIN,HD) row-major -> WT (HD,CIN) f16 so B-frags are k-contiguous
__global__ void k_cast_winT(const float* __restrict__ w, _Float16* __restrict__ d) {
  int i = blockIdx.x * 256 + threadIdx.x;
  if (i >= HD * CIN) return;
  int o = i / CIN, k = i % CIN;
  d[i] = (_Float16)w[k * HD + o];
}

// ---------------- per-layer SSM params: w = exp(dt*A), c = 2*C*(w-1)/A ----------------
__global__ void k_params(const float* __restrict__ log_dt, const float* __restrict__ Cri,
                         const float* __restrict__ lAr, const float* __restrict__ Ai,
                         float4* __restrict__ P) {
  int t = blockIdx.x * 256 + threadIdx.x;
  if (t >= HD * NST) return;
  int h = t / NST;
  float dt  = expf(log_dt[h]);
  float Ar  = -expf(lAr[t]);
  float Aim = Ai[t];
  float Crr = Cri[2 * t], Cii = Cri[2 * t + 1];
  float er = expf(Ar * dt);
  float sn, cs;
  sincosf(Aim * dt, &sn, &cs);
  float wr = er * cs, wi = er * sn;
  float inv = 1.f / (Ar * Ar + Aim * Aim);
  float xr = wr - 1.f, xi = wi;
  float qr = (xr * Ar + xi * Aim) * inv;   // (w-1)/A
  float qi = (xi * Ar - xr * Aim) * inv;
  float Ctr = Crr * qr - Cii * qi;
  float Cti = Crr * qi + Cii * qr;
  P[t] = make_float4(wr, wi, 2.f * Ctr, 2.f * Cti);
}

// ---------------- SSM chunked scan: pass 1 (chunk-final states) ----------------
__global__ void __launch_bounds__(256) k_scan1(const float* __restrict__ U,
                                               const float4* __restrict__ P,
                                               float2* __restrict__ F) {
  int idx = blockIdx.x * 256 + threadIdx.x;
  if (idx >= Bsz * HD * (NCH - 1)) return;
  int h = idx % HD;
  int t = idx / HD;
  int c = t % (NCH - 1);
  int b = t / (NCH - 1);
  float wr[NST], wi[NST], sr[NST], si[NST];
  const float4* p = P + h * NST;
#pragma unroll
  for (int n = 0; n < NST; ++n) { float4 q = p[n]; wr[n] = q.x; wi[n] = q.y; sr[n] = 0.f; si[n] = 0.f; }
  const float* up = U + ((size_t)(b * Lseq + c * CLEN)) * HD + h;
  for (int l = 0; l < CLEN; ++l) {
    float u = *up; up += HD;
#pragma unroll
    for (int n = 0; n < NST; ++n) {
      float nsr = fmaf(wr[n], sr[n], fmaf(-wi[n], si[n], u));
      float nsi = fmaf(wr[n], si[n], wi[n] * sr[n]);
      sr[n] = nsr; si[n] = nsi;
    }
  }
  float2* f = F + ((size_t)(b * (NCH - 1) + c)) * NST * HD + h;
#pragma unroll
  for (int n = 0; n < NST; ++n) f[n * HD] = make_float2(sr[n], si[n]);
}

// ---------------- pass 2: carry (Horner with w^CLEN) + outputs + D skip + GELU ----------------
__global__ void __launch_bounds__(256) k_scan2(const float* __restrict__ U,
                                               const float4* __restrict__ P,
                                               const float2* __restrict__ F,
                                               const float* __restrict__ Dv,
                                               _Float16* __restrict__ Y) {
  int idx = blockIdx.x * 256 + threadIdx.x;
  if (idx >= Bsz * HD * NCH) return;
  int h = idx % HD;
  int t = idx / HD;
  int c = t % NCH;
  int b = t / NCH;
  float wr[NST], wi[NST], cr[NST], ci[NST], sr[NST], si[NST];
  const float4* p = P + h * NST;
#pragma unroll
  for (int n = 0; n < NST; ++n) {
    float4 q = p[n];
    wr[n] = q.x; wi[n] = q.y; cr[n] = q.z; ci[n] = q.w;
    sr[n] = 0.f; si[n] = 0.f;
  }
  if (c > 0) {
    float pr[NST], pi[NST];
#pragma unroll
    for (int n = 0; n < NST; ++n) { pr[n] = wr[n]; pi[n] = wi[n]; }
    for (int s = 0; s < 8; ++s) {  // w^(2^8) = w^CLEN
#pragma unroll
      for (int n = 0; n < NST; ++n) {
        float r = pr[n] * pr[n] - pi[n] * pi[n];
        float m = 2.f * pr[n] * pi[n];
        pr[n] = r; pi[n] = m;
      }
    }
    for (int k = 0; k < c; ++k) {   // carry = Horner over chunk-final states
      const float2* f = F + ((size_t)(b * (NCH - 1) + k)) * NST * HD + h;
#pragma unroll
      for (int n = 0; n < NST; ++n) {
        float2 fk = f[n * HD];
        float nsr = fmaf(pr[n], sr[n], fmaf(-pi[n], si[n], fk.x));
        float nsi = fmaf(pr[n], si[n], fmaf(pi[n], sr[n], fk.y));
        sr[n] = nsr; si[n] = nsi;
      }
    }
  }
  float dh = Dv[h];
  const float* up = U + ((size_t)(b * Lseq + c * CLEN)) * HD + h;
  _Float16* yp = Y + ((size_t)(b * Lseq + c * CLEN)) * HD + h;
  for (int l = 0; l < CLEN; ++l) {
    float u = *up; up += HD;
    float acc = 0.f;
#pragma unroll
    for (int n = 0; n < NST; ++n) {
      float nsr = fmaf(wr[n], sr[n], fmaf(-wi[n], si[n], u));
      float nsi = fmaf(wr[n], si[n], wi[n] * sr[n]);
      sr[n] = nsr; si[n] = nsi;
      acc = fmaf(cr[n], nsr, fmaf(-ci[n], nsi, acc));
    }
    float y = fmaf(u, dh, acc);
    y = 0.5f * y * (1.f + erff(y * 0.70710678118654752f));  // exact-erf GELU
    *yp = (_Float16)y; yp += HD;
  }
}

// ---------------- input GEMM: U = X16 @ WinT^T + b_in  (M=B*L, K=192, N=512) ----------------
__global__ void __launch_bounds__(256) k_gemm_in(const _Float16* __restrict__ X,
                                                 const _Float16* __restrict__ WT,
                                                 const float* __restrict__ bin,
                                                 float* __restrict__ Uo) {
  __shared__ __align__(16) _Float16 As[2][128][40];   // double-buffered A tile
  int tid = threadIdx.x;
  int wave = tid >> 5, lane = tid & 31;
  int hi = lane >> 4, ln = lane & 15;
  int m0 = blockIdx.x * 128, n0 = blockIdx.y * 64;
  int mw = (wave & 1) * 64, nw = (wave >> 1) * 16;
  v8f z = {0.f, 0.f, 0.f, 0.f, 0.f, 0.f, 0.f, 0.f};
  v8f acc[4] = {z, z, z, z};
  const _Float16* wa = WT + (size_t)(n0 + nw + ln) * CIN;
  int r = tid >> 1, cc = (tid & 1) * 16;
  const _Float16* gsrc = X + (size_t)(m0 + r) * CIN + cc;
  // prologue: chunk 0 -> buffer 0
  async_cp16(&As[0][r][cc],     gsrc);
  async_cp16(&As[0][r][cc] + 8, gsrc + 8);
  const int NCHUNK = CIN / 32;
  for (int i = 0; i < NCHUNK; ++i) {
    int k0 = i * 32;
    if (i + 1 < NCHUNK) {   // prefetch chunk i+1 into alternate buffer
      _Float16* nb = &As[(i + 1) & 1][r][cc];
      async_cp16(nb,     gsrc + k0 + 32);
      async_cp16(nb + 8, gsrc + k0 + 40);
      async_wait_le2();     // oldest 2 (chunk i) have landed; chunk i+1 still in flight
    } else {
      async_wait_le0();
    }
    __syncthreads();
    v16h bf = b_frag(wa + k0, hi);
#pragma unroll
    for (int s = 0; s < 4; ++s) {
      v16h af = a_frag(&As[i & 1][mw + s * 16 + ln][0], hi);
      acc[s] = wmma16(af, bf, acc[s]);
    }
    __syncthreads();        // all reads of buffer (i&1) done before it is refilled at i+2
  }
  int col = n0 + nw + ln;
  float bias = bin[col];
#pragma unroll
  for (int s = 0; s < 4; ++s)
#pragma unroll
    for (int rr = 0; rr < 8; ++rr) {
      int row = m0 + mw + s * 16 + rr + 8 * hi;
      Uo[(size_t)row * HD + col] = acc[s][rr] + bias;
    }
}

// ---------------- conv1x1 + fused GLU: U = glu(Y16 @ W^T + cb)  (K=512, 512 out pairs) ----------------
__global__ void __launch_bounds__(256) k_conv_glu(const _Float16* __restrict__ Y,
                                                  const _Float16* __restrict__ W,
                                                  const float* __restrict__ cb,
                                                  float* __restrict__ Uo) {
  __shared__ __align__(16) _Float16 As[2][128][40];   // double-buffered A tile
  int tid = threadIdx.x;
  int wave = tid >> 5, lane = tid & 31;
  int hi = lane >> 4, ln = lane & 15;
  int m0 = blockIdx.x * 128, n0 = blockIdx.y * 64;
  int mw = (wave & 1) * 64, nw = (wave >> 1) * 16;
  v8f z = {0.f, 0.f, 0.f, 0.f, 0.f, 0.f, 0.f, 0.f};
  v8f accA[4] = {z, z, z, z};
  v8f accB[4] = {z, z, z, z};
  const _Float16* wa = W + (size_t)(n0 + nw + ln) * HD;        // 'a' half: row o
  const _Float16* wb = W + (size_t)(HD + n0 + nw + ln) * HD;   // 'b' half: row o+512
  int r = tid >> 1, cc = (tid & 1) * 16;
  const _Float16* gsrc = Y + (size_t)(m0 + r) * HD + cc;
  // prologue: chunk 0 -> buffer 0
  async_cp16(&As[0][r][cc],     gsrc);
  async_cp16(&As[0][r][cc] + 8, gsrc + 8);
  const int NCHUNK = HD / 32;
  for (int i = 0; i < NCHUNK; ++i) {
    int k0 = i * 32;
    if (i + 1 < NCHUNK) {   // prefetch chunk i+1 into alternate buffer
      _Float16* nb = &As[(i + 1) & 1][r][cc];
      async_cp16(nb,     gsrc + k0 + 32);
      async_cp16(nb + 8, gsrc + k0 + 40);
      async_wait_le2();
    } else {
      async_wait_le0();
    }
    __syncthreads();
    v16h ba = b_frag(wa + k0, hi);
    v16h bb = b_frag(wb + k0, hi);
#pragma unroll
    for (int s = 0; s < 4; ++s) {
      v16h af = a_frag(&As[i & 1][mw + s * 16 + ln][0], hi);
      accA[s] = wmma16(af, ba, accA[s]);
      accB[s] = wmma16(af, bb, accB[s]);
    }
    __syncthreads();
  }
  int col = n0 + nw + ln;
  float biasA = cb[col], biasB = cb[HD + col];
#pragma unroll
  for (int s = 0; s < 4; ++s)
#pragma unroll
    for (int rr = 0; rr < 8; ++rr) {
      float a  = accA[s][rr] + biasA;
      float bg = accB[s][rr] + biasB;
      float o  = a * (1.f / (1.f + expf(-bg)));   // a * sigmoid(b)
      int row = m0 + mw + s * 16 + rr + 8 * hi;
      Uo[(size_t)row * HD + col] = o;
    }
}

// ---------------- tiny output GEMM (N=2): plain f32 ----------------
__global__ void k_gemm_out(const float* __restrict__ U, const float* __restrict__ Wo,
                           const float* __restrict__ bo, float* __restrict__ out) {
  int m = blockIdx.x * 256 + threadIdx.x;
  if (m >= Bsz * Lseq) return;
  const float* row = U + (size_t)m * HD;
  float a0 = 0.f, a1 = 0.f;
  for (int k = 0; k < HD; k += 4) {
    float4 v = *(const float4*)(row + k);
    a0 = fmaf(v.x, Wo[2 * k + 0], a0); a1 = fmaf(v.x, Wo[2 * k + 1], a1);
    a0 = fmaf(v.y, Wo[2 * k + 2], a0); a1 = fmaf(v.y, Wo[2 * k + 3], a1);
    a0 = fmaf(v.z, Wo[2 * k + 4], a0); a1 = fmaf(v.z, Wo[2 * k + 5], a1);
    a0 = fmaf(v.w, Wo[2 * k + 6], a0); a1 = fmaf(v.w, Wo[2 * k + 7], a1);
  }
  out[2 * m]     = a0 + bo[0];
  out[2 * m + 1] = a1 + bo[1];
}

// ---------------- host launcher ----------------
extern "C" void kernel_launch(void* const* d_in, const int* in_sizes, int n_in,
                              void* d_out, int out_size, void* d_ws, size_t ws_size,
                              hipStream_t stream) {
  const float* neural_input = (const float*)d_in[0];
  const float* W_in   = (const float*)d_in[1];
  const float* b_in   = (const float*)d_in[2];
  const float* log_dt = (const float*)d_in[3];
  const float* Cri    = (const float*)d_in[4];
  const float* lAr    = (const float*)d_in[5];
  const float* Aim    = (const float*)d_in[6];
  const float* Dp     = (const float*)d_in[7];
  const float* conv_w = (const float*)d_in[8];
  const float* conv_b = (const float*)d_in[9];
  const float* W_out  = (const float*)d_in[10];
  const float* b_out  = (const float*)d_in[11];

  char* ws = (char*)d_ws;
  auto alloc = [&](size_t bytes) -> char* {
    char* p = ws;
    ws += (bytes + 255) & ~(size_t)255;
    return p;
  };
  float*    U    = (float*)alloc(sizeof(float) * Bsz * Lseq * HD);          // 128 MB ping
  _Float16* Y16  = (_Float16*)alloc(sizeof(_Float16) * Bsz * Lseq * HD);    // 64 MB
  _Float16* X16  = (_Float16*)alloc(sizeof(_Float16) * Bsz * Lseq * CIN);   // 24 MB
  _Float16* WT16 = (_Float16*)alloc(sizeof(_Float16) * HD * CIN);
  _Float16* Wc16 = (_Float16*)alloc(sizeof(_Float16) * NLAY * 2 * HD * HD); // 4 MB
  float4*   P    = (float4*)alloc(sizeof(float4) * HD * NST);
  float2*   F    = (float2*)alloc(sizeof(float2) * Bsz * (NCH - 1) * NST * HD); // ~29 MB

  int nX = Bsz * Lseq * CIN;
  k_cast_f16<<<(nX + 255) / 256, 256, 0, stream>>>(neural_input, X16, nX);
  k_cast_winT<<<(HD * CIN + 255) / 256, 256, 0, stream>>>(W_in, WT16);
  int nW = NLAY * 2 * HD * HD;
  k_cast_f16<<<(nW + 255) / 256, 256, 0, stream>>>(conv_w, Wc16, nW);

  k_gemm_in<<<dim3((Bsz * Lseq) / 128, HD / 64), 256, 0, stream>>>(X16, WT16, b_in, U);

  for (int l = 0; l < NLAY; ++l) {
    k_params<<<(HD * NST + 255) / 256, 256, 0, stream>>>(
        log_dt + (size_t)l * HD, Cri + (size_t)l * HD * NST * 2,
        lAr + (size_t)l * HD * NST, Aim + (size_t)l * HD * NST, P);
    int n1 = Bsz * HD * (NCH - 1);
    k_scan1<<<(n1 + 255) / 256, 256, 0, stream>>>(U, P, F);
    int n2 = Bsz * HD * NCH;
    k_scan2<<<(n2 + 255) / 256, 256, 0, stream>>>(U, P, F, Dp + (size_t)l * HD, Y16);
    k_conv_glu<<<dim3((Bsz * Lseq) / 128, HD / 64), 256, 0, stream>>>(
        Y16, Wc16 + (size_t)l * 2 * HD * HD, conv_b + (size_t)l * 2 * HD, U);
  }

  k_gemm_out<<<(Bsz * Lseq + 255) / 256, 256, 0, stream>>>(U, W_out, b_out, (float*)d_out);
}
